// GCN_28922309771311
// MI455X (gfx1250) — matrix operand order repaired and checked by reading, not compiled, not run on hardware
//
#include <hip/hip_runtime.h>

// Problem constants (fixed by the reference)
#define NF 128   // node features == hidden
#define NH 128
#define NC 64    // classes
#define NG 64    // graphs

typedef __attribute__((ext_vector_type(16))) __bf16 v16bf;
typedef __attribute__((ext_vector_type(8)))  float  v8f;
typedef __attribute__((ext_vector_type(4)))  __bf16 bf4;

union Frag { v16bf v; uint4 q[2]; };

// ---------------------------------------------------------------------------
// f32 -> bf16 weight conversion (one-time per launch, 16384 elems per matrix)
// ---------------------------------------------------------------------------
__global__ void cvt_bf16_kernel(const float* __restrict__ in, __bf16* __restrict__ out, int n) {
    int i = blockIdx.x * blockDim.x + threadIdx.x;
    if (i < n) out[i] = (__bf16)in[i];
}

// ---------------------------------------------------------------------------
// Edge scatter: one wave32 per edge, lane handles 4 consecutive floats.
// agg[dst] += h[src]; optionally deg[dst] += 1.
// ---------------------------------------------------------------------------
__global__ void scatter_kernel(const float* __restrict__ h,
                               const long long* __restrict__ src,
                               const long long* __restrict__ dst,
                               float* __restrict__ agg, float* __restrict__ deg,
                               int E, int computeDeg) {
    int gid  = blockIdx.x * blockDim.x + threadIdx.x;
    int e    = gid >> 5;
    int lane = gid & 31;
    if (e >= E) return;
    long long s = src[e], d = dst[e];
    const float4 v = ((const float4*)(h + (size_t)s * NF))[lane];
    float* ap = agg + (size_t)d * NF + lane * 4;
    atomicAdd(ap + 0, v.x);
    atomicAdd(ap + 1, v.y);
    atomicAdd(ap + 2, v.z);
    atomicAdd(ap + 3, v.w);
    if (computeDeg && lane == 0) atomicAdd(deg + d, 1.0f);
}

// ---------------------------------------------------------------------------
// Fused SAGEConv GEMM: out = relu?( (agg/deg) @ Wl^T + bl + x @ Wr^T )
// Block = 256 threads (8 waves), tile = 128 rows x 128 cols, K = 128.
// bf16 WMMA (16x16x32), f32 accumulate. A tiles staged in LDS (pad stride 136
// bf16 = 272B: 16B-aligned, conflict-free for 16-lane frag loads). B frags
// streamed from L2-resident bf16 weights (K-contiguous per ISA B layout).
// Epilogue: full tiles (781/782 blocks) take an unguarded store path; the
// uniform relu flag is branched once per wave (SGPR), not selected per lane.
// ---------------------------------------------------------------------------
__global__ __launch_bounds__(256)
void sage_gemm_kernel(const float* __restrict__ agg, const float* __restrict__ deg,
                      const float* __restrict__ xin,
                      const __bf16* __restrict__ Wl, const float* __restrict__ bl,
                      const __bf16* __restrict__ Wr,
                      float* __restrict__ out, int n, int do_relu) {
    __shared__ __bf16 ldsA[128 * 136];   // scaled agg tile (bf16)
    __shared__ __bf16 ldsX[128 * 136];   // x tile (bf16)

    const int t    = threadIdx.x;
    const int row0 = blockIdx.x * 128;

    // ---- stage: 2 threads per row, each converts 64 floats (16 x float4) ----
    {
        int  r    = t >> 1;
        int  col0 = (t & 1) * 64;
        int  gr   = row0 + r;
        bool valid = gr < n;
        float rd = 0.f;
        if (valid) rd = 1.0f / fmaxf(deg[gr], 1.0f);
        const float4* pa = (const float4*)(agg + (size_t)gr * NF + col0);
        const float4* px = (const float4*)(xin + (size_t)gr * NF + col0);
        __bf16* la = ldsA + r * 136 + col0;
        __bf16* lx = ldsX + r * 136 + col0;
        float4 z = make_float4(0.f, 0.f, 0.f, 0.f);
        #pragma unroll
        for (int i = 0; i < 16; ++i) {
            float4 va = valid ? pa[i] : z;
            float4 vx = valid ? px[i] : z;
            bf4 ba = { (__bf16)(va.x * rd), (__bf16)(va.y * rd),
                       (__bf16)(va.z * rd), (__bf16)(va.w * rd) };
            bf4 bx = { (__bf16)vx.x, (__bf16)vx.y, (__bf16)vx.z, (__bf16)vx.w };
            *(bf4*)(la + i * 4) = ba;
            *(bf4*)(lx + i * 4) = bx;
        }
    }
    __syncthreads();

    const int wave      = t >> 5;
    const int lane      = t & 31;
    const int m0        = wave * 16;      // 16-row strip per wave
    const int rowInTile = lane & 15;
    const int hi        = lane >> 4;      // lane half (0/1)

    // ---- hoist A fragments: 4 k-steps x {agg, x} (ISA 16-bit A layout) ----
    Frag fa[4], fx[4];
    const __bf16* baseA = ldsA + (m0 + rowInTile) * 136;
    const __bf16* baseX = ldsX + (m0 + rowInTile) * 136;
    #pragma unroll
    for (int ks = 0; ks < 4; ++ks) {
        int kb = ks * 32 + hi * 8;                       // K 0-7 / 8-15 halves
        fa[ks].q[0] = *(const uint4*)(baseA + kb);
        fa[ks].q[1] = *(const uint4*)(baseA + kb + 16);  // K 16-23 / 24-31
        fx[ks].q[0] = *(const uint4*)(baseX + kb);
        fx[ks].q[1] = *(const uint4*)(baseX + kb + 16);
    }

    const bool fullTile = (row0 + 128) <= n;     // block-uniform
    const int  rbase    = row0 + m0 + hi * 8;    // C/D layout M base for this lane

    #pragma unroll
    for (int nt = 0; nt < 8; ++nt) {
        int   col = nt * 16 + rowInTile;   // output column == weight row (B col)
        float bv  = bl[col];
        v8f acc = { bv, bv, bv, bv, bv, bv, bv, bv };
        #pragma unroll
        for (int ks = 0; ks < 4; ++ks) {
            int kb2 = ks * 32 + hi * 8;    // B layout: K 0-7 / 8-15 halves, then +16
            Frag bWl, bWr;
            const uint4* pl = (const uint4*)(Wl + col * NF + kb2);
            bWl.q[0] = pl[0]; bWl.q[1] = pl[2];
            const uint4* pr = (const uint4*)(Wr + col * NF + kb2);
            bWr.q[0] = pr[0]; bWr.q[1] = pr[2];
            acc = __builtin_amdgcn_wmma_f32_16x16x32_bf16(
                      false, fa[ks].v, false, bWl.v, (short)0, acc, false, false);
            acc = __builtin_amdgcn_wmma_f32_16x16x32_bf16(
                      false, fx[ks].v, false, bWr.v, (short)0, acc, false, false);
        }
        // C/D layout: VGPR r -> M = rbase + r, N = col
        float* po = out + (size_t)rbase * NH + col;
        if (fullTile) {
            if (do_relu) {
                #pragma unroll
                for (int r = 0; r < 8; ++r) po[(size_t)r * NH] = fmaxf(acc[r], 0.f);
            } else {
                #pragma unroll
                for (int r = 0; r < 8; ++r) po[(size_t)r * NH] = acc[r];
            }
        } else {
            #pragma unroll
            for (int r = 0; r < 8; ++r) {
                if (rbase + r < n) {
                    float v = acc[r];
                    if (do_relu) v = fmaxf(v, 0.f);
                    po[(size_t)r * NH] = v;
                }
            }
        }
    }
}

// ---------------------------------------------------------------------------
// Pooling: batch is sorted, so accumulate per-run locally and flush atomics
// only on graph-id change. Float max via monotone u32 encoding.
// ---------------------------------------------------------------------------
__device__ __forceinline__ unsigned encf(float x) {
    unsigned u = __float_as_uint(x);
    return ((int)u < 0) ? ~u : (u | 0x80000000u);
}
__device__ __forceinline__ float decf(unsigned u) {
    return __uint_as_float((u & 0x80000000u) ? (u & 0x7FFFFFFFu) : ~u);
}

__global__ void pool_kernel(const float* __restrict__ h, const long long* __restrict__ batch,
                            float* __restrict__ gsum, unsigned* __restrict__ gmax,
                            float* __restrict__ gcnt, int n) {
    const int f   = threadIdx.x & 127;          // feature lane (coalesced rows)
    const int grp = threadIdx.x >> 7;           // 0..1
    const int CH  = 32;                         // nodes per group
    int n0 = (blockIdx.x * 2 + grp) * CH;
    float ls = 0.f, lm = -3.402823466e38f, lc = 0.f;
    long long cur = -1;
    for (int i = 0; i < CH; ++i) {
        int node = n0 + i;
        if (node >= n) break;
        long long g = batch[node];
        if (g != cur) {
            if (cur >= 0) {
                atomicAdd(&gsum[cur * 128 + f], ls);
                atomicMax(&gmax[cur * 128 + f], encf(lm));
                if (f == 0) atomicAdd(&gcnt[cur], lc);
            }
            cur = g; ls = 0.f; lm = -3.402823466e38f; lc = 0.f;
        }
        float v = h[(size_t)node * 128 + f];
        ls += v; lm = fmaxf(lm, v); lc += 1.f;
    }
    if (cur >= 0) {
        atomicAdd(&gsum[cur * 128 + f], ls);
        atomicMax(&gmax[cur * 128 + f], encf(lm));
        if (f == 0) atomicAdd(&gcnt[cur], lc);
    }
}

__global__ void pool_finalize_kernel(const float* __restrict__ gsum,
                                     const unsigned* __restrict__ gmax,
                                     const float* __restrict__ gcnt,
                                     float* __restrict__ pooled) {
    int idx = blockIdx.x * blockDim.x + threadIdx.x;
    if (idx >= NG * 128) return;
    int g = idx >> 7, f = idx & 127;
    float c    = gcnt[g];
    float mean = gsum[idx] / fmaxf(c, 1.0f);
    float mx   = 0.f;
    if (c > 0.f) {
        mx = decf(gmax[idx]);
        if (!(mx > -3.4e38f && mx < 3.4e38f)) mx = 0.f;  // isfinite guard
    }
    pooled[g * 256 + f]       = mx;    // [max | mean] concat, axis 1
    pooled[g * 256 + 128 + f] = mean;
}

// ---------------------------------------------------------------------------
// Head: out[g][c] = blin[c] + pooled[g,:] . Wlin[c,:]   (64x256x64, trivial)
// ---------------------------------------------------------------------------
__global__ void linear_kernel(const float* __restrict__ pooled,
                              const float* __restrict__ W,
                              const float* __restrict__ b,
                              float* __restrict__ out) {
    int idx = blockIdx.x * blockDim.x + threadIdx.x;
    if (idx >= NG * NC) return;
    int g = idx >> 6, c = idx & 63;
    float acc = b[c];
    const float4* pp = (const float4*)(pooled + g * 256);
    const float4* pw = (const float4*)(W + c * 256);
    #pragma unroll 4
    for (int k = 0; k < 64; ++k) {
        float4 a = pp[k], w = pw[k];
        acc += a.x * w.x + a.y * w.y + a.z * w.z + a.w * w.w;
    }
    out[idx] = acc;
}

// ---------------------------------------------------------------------------
extern "C" void kernel_launch(void* const* d_in, const int* in_sizes, int n_in,
                              void* d_out, int out_size, void* d_ws, size_t ws_size,
                              hipStream_t stream) {
    const float*     x     = (const float*)d_in[0];
    const long long* ei    = (const long long*)d_in[1];   // int64 [2, E]
    const long long* batch = (const long long*)d_in[2];   // int64 [N]
    const float* Wl1 = (const float*)d_in[3];
    const float* bl1 = (const float*)d_in[4];
    const float* Wr1 = (const float*)d_in[5];
    const float* Wl2 = (const float*)d_in[6];
    const float* bl2 = (const float*)d_in[7];
    const float* Wr2 = (const float*)d_in[8];
    const float* Wl3 = (const float*)d_in[9];
    const float* bl3 = (const float*)d_in[10];
    const float* Wr3 = (const float*)d_in[11];
    const float* Wlin = (const float*)d_in[12];
    const float* blin = (const float*)d_in[13];

    const int N = in_sizes[2];
    const int E = in_sizes[1] / 2;
    const long long* src = ei;
    const long long* dst = ei + E;

    // workspace carve-up (256B aligned)
    char* ws = (char*)d_ws;
    size_t off = 0;
    auto alloc = [&](size_t bytes) -> void* {
        void* p = ws + off;
        off = (off + bytes + 255) & ~(size_t)255;
        return p;
    };
    float*    agg  = (float*)alloc((size_t)N * NF * 4);
    float*    hA   = (float*)alloc((size_t)N * NF * 4);
    float*    hB   = (float*)alloc((size_t)N * NF * 4);
    float*    deg  = (float*)alloc((size_t)N * 4);
    __bf16*   wb[6];
    for (int i = 0; i < 6; ++i) wb[i] = (__bf16*)alloc((size_t)NF * NH * 2);
    float*    gsum = (float*)alloc((size_t)NG * 128 * 4);
    unsigned* gmax = (unsigned*)alloc((size_t)NG * 128 * 4);
    float*    gcnt = (float*)alloc((size_t)NG * 4);
    float*    pooled = (float*)alloc((size_t)NG * 256 * 4);

    // one-time bf16 weight conversion (order: Wl1, Wr1, Wl2, Wr2, Wl3, Wr3)
    const float* wsrc[6] = { Wl1, Wr1, Wl2, Wr2, Wl3, Wr3 };
    for (int i = 0; i < 6; ++i)
        cvt_bf16_kernel<<<(NF * NH + 255) / 256, 256, 0, stream>>>(wsrc[i], wb[i], NF * NH);

    hipMemsetAsync(deg, 0, (size_t)N * 4, stream);

    dim3 blk(256);
    dim3 grdS((unsigned)((E * 32 + 255) / 256));
    dim3 grdG((unsigned)((N + 127) / 128));
    dim3 grdP((unsigned)((N + 63) / 64));

    // layer 1 (relu)
    hipMemsetAsync(agg, 0, (size_t)N * NF * 4, stream);
    scatter_kernel<<<grdS, blk, 0, stream>>>(x, src, dst, agg, deg, E, 1);
    sage_gemm_kernel<<<grdG, blk, 0, stream>>>(agg, deg, x, wb[0], bl1, wb[1], hA, N, 1);
    // layer 2 (relu)
    hipMemsetAsync(agg, 0, (size_t)N * NF * 4, stream);
    scatter_kernel<<<grdS, blk, 0, stream>>>(hA, src, dst, agg, deg, E, 0);
    sage_gemm_kernel<<<grdG, blk, 0, stream>>>(agg, deg, hA, wb[2], bl2, wb[3], hB, N, 1);
    // layer 3 (no relu)
    hipMemsetAsync(agg, 0, (size_t)N * NF * 4, stream);
    scatter_kernel<<<grdS, blk, 0, stream>>>(hB, src, dst, agg, deg, E, 0);
    sage_gemm_kernel<<<grdG, blk, 0, stream>>>(agg, deg, hB, wb[4], bl3, wb[5], hA, N, 0);

    // pooling + head
    hipMemsetAsync(gsum, 0, (size_t)NG * 128 * 4, stream);
    hipMemsetAsync(gmax, 0, (size_t)NG * 128 * 4, stream);  // 0 == enc(-inf) floor
    hipMemsetAsync(gcnt, 0, (size_t)NG * 4, stream);
    pool_kernel<<<grdP, blk, 0, stream>>>(hA, batch, gsum, gmax, gcnt, N);
    pool_finalize_kernel<<<(NG * 128 + 255) / 256, 256, 0, stream>>>(gsum, gmax, gcnt, pooled);
    linear_kernel<<<(NG * NC + 255) / 256, 256, 0, stream>>>(pooled, Wlin, blin, (float*)d_out);
}
